// MultiheadAttentionWithGate_66795331387844
// MI455X (gfx1250) — matrix operand order repaired and checked by reading, not compiled, not run on hardware
//
#include <hip/hip_runtime.h>

typedef __attribute__((ext_vector_type(16))) __bf16 v16bf;
typedef __attribute__((ext_vector_type(8)))  __bf16 v8bf;
typedef __attribute__((ext_vector_type(8)))  float  v8f;
typedef __attribute__((ext_vector_type(4)))  int    v4i;

#define B_  4
#define S_  1024
#define E_  1024
#define H_  16
#define D_  64

// -------- gfx1250 async global->LDS staging (ASYNCcnt path), guarded --------
#if __has_builtin(__builtin_amdgcn_global_load_async_to_lds_b128) && \
    __has_builtin(__builtin_amdgcn_s_wait_asynccnt)
#define HAVE_ASYNC_LDS 1
#endif

__device__ __forceinline__ void stage16(const __bf16* g, __bf16* l) {
#ifdef HAVE_ASYNC_LDS
  __builtin_amdgcn_global_load_async_to_lds_b128(
      (__attribute__((address_space(1))) v4i*)g,
      (__attribute__((address_space(3))) v4i*)l, 0, 0);
#else
  *(v8bf*)l = *(const v8bf*)g;
#endif
}
__device__ __forceinline__ void stage_wait() {
#ifdef HAVE_ASYNC_LDS
  __builtin_amdgcn_s_wait_asynccnt(0);
#endif
}

// ---------------- WMMA fragment helpers (CDNA5 16x16x32 bf16 layouts) ----------------
// A 16x32 bf16 (ISA 7.12.2): lanes 0-15 -> M=lane, K groups {0..7,16..23};
//                            lanes 16-31 -> M=lane-16, K groups {8..15,24..31}; 2 bf16/VGPR.
__device__ __forceinline__ v16bf frag_a(const __bf16* base, int ld, int m0, int k0, int lane) {
  v16bf f;
  int m  = m0 + (lane & 15);
  int kb = (lane & 16) ? 8 : 0;
  const __bf16* p = base + m * ld + k0 + kb;
#pragma unroll
  for (int v = 0; v < 4; ++v) { f[2*v] = p[2*v]; f[2*v+1] = p[2*v+1]; }
  p += 16;
#pragma unroll
  for (int v = 0; v < 4; ++v) { f[8+2*v] = p[2*v]; f[9+2*v] = p[2*v+1]; }
  return f;
}

// B 32x16 bf16: lane = column (lanes 0-15 K={0..7,16..23}, lanes 16-31 K={8..15,24..31}).
// Staged n-major in LDS: base[n*ld + k].
__device__ __forceinline__ v16bf frag_b(const __bf16* base, int ld, int n0, int k0, int lane) {
  v16bf f;
  int n  = n0 + (lane & 15);
  int kb = (lane & 16) ? 8 : 0;
  const __bf16* p = base + n * ld + k0 + kb;
#pragma unroll
  for (int t = 0; t < 16; ++t) f[t] = p[t + (t & 8)];
  return f;
}

#define WMMA_BF16(A, Bm, C) \
  __builtin_amdgcn_wmma_f32_16x16x32_bf16(false, (A), false, (Bm), (short)0, (C), false, false)

// ---------------- K0: f32 -> bf16 convert ----------------
__global__ void cvt_f32_bf16(const float* __restrict__ src, __bf16* __restrict__ dst, int n) {
  int i = blockIdx.x * blockDim.x + threadIdx.x;
  int stride = gridDim.x * blockDim.x;
  for (; i < n; i += stride) dst[i] = (__bf16)src[i];
}

// ---------------- K1: in-projection GEMM (4096 x 3072, K=1024) ----------------
__global__ __launch_bounds__(128) void inproj_gemm(
    const __bf16* __restrict__ xq, const __bf16* __restrict__ xk, const __bf16* __restrict__ xv,
    const __bf16* __restrict__ w,  const float* __restrict__ bias,
    __bf16* __restrict__ qh, __bf16* __restrict__ kh, __bf16* __restrict__ vh) {
  __shared__ __bf16 As[64][72];   // 64 rows x 64 k (+8 pad), 9216 B each
  __shared__ __bf16 Bs[64][72];
  int m0 = blockIdx.x * 64;
  int n0 = blockIdx.y * 64;
  int proj = n0 >> 10;
  const __bf16* A = (proj == 0) ? xq : ((proj == 1) ? xk : xv);
  int tid = threadIdx.x, lane = tid & 31, wave = tid >> 5;

  v8f acc[4] = {};
  for (int k0 = 0; k0 < E_; k0 += 64) {
    __syncthreads();
    for (int c = tid; c < 512; c += 128) {       // 64 rows x 8 chunks of 16B
      int r = c >> 3, seg = c & 7;
      stage16(&A[(m0 + r) * E_ + k0 + seg * 8], &As[r][seg * 8]);
      stage16(&w[(n0 + r) * E_ + k0 + seg * 8], &Bs[r][seg * 8]);
    }
    if (k0 + 128 < E_)  // gfx1250 global_prefetch of tile after next
      __builtin_prefetch(&A[(m0 + (tid >> 1)) * E_ + k0 + 128], 0, 1);
    stage_wait();
    __syncthreads();
#pragma unroll
    for (int ko = 0; ko < 2; ++ko) {
      v16bf af = frag_a(&As[0][0], 72, wave * 16, ko * 32, lane);
#pragma unroll
      for (int t = 0; t < 4; ++t) {
        v16bf bf = frag_b(&Bs[0][0], 72, t * 16, ko * 32, lane);
        acc[t] = WMMA_BF16(af, bf, acc[t]);
      }
    }
  }
  // Epilogue: C/D layout -> lane = column, VGPR idx = row (+8 for upper lanes)
  int col16 = lane & 15, rbase = (lane & 16) ? 8 : 0;
  float scale = (proj == 0) ? 0.125f : 1.0f;
#pragma unroll
  for (int t = 0; t < 4; ++t) {
    int f = n0 + t * 16 + col16;
    float bv = bias[f];
    int fh = f & 1023, h = fh >> 6, d = fh & 63;
    __bf16* dst = (proj == 0) ? qh : ((proj == 1) ? kh : vh);
#pragma unroll
    for (int r = 0; r < 8; ++r) {
      int row = m0 + wave * 16 + rbase + r;      // b*S + s
      int b = row >> 10, s = row & 1023;
      float y = (acc[t][r] + bv) * scale;
      dst[(size_t)((b * H_ + h) * S_ + s) * D_ + d] = (__bf16)y;
    }
  }
}

// ---------------- K2: fused attention + head-avg + gate ----------------
// 1 block = (b, 16-row q tile), 8 waves; wave w owns key range [w*128, w*128+128)
__global__ __launch_bounds__(256) void attn_kernel(
    const __bf16* __restrict__ qh, const __bf16* __restrict__ kh, const __bf16* __restrict__ vh,
    const float* __restrict__ gate, __bf16* __restrict__ ctxw, float* __restrict__ gated_out) {
  __shared__ __bf16 qs[16][1032];          // q rows for all heads (e = h*64+d)
  __shared__ __bf16 kvs[8][64][64];        // per-wave K (j,d) or V^T (d,j) slab
  __shared__ __bf16 ps[8][16][128];        // per-wave probs in A layout (m, j_local)
  __shared__ float  redbuf[8][16];
  __shared__ float  ctxred[16][64];

  int bqt = blockIdx.x;
  int b = bqt >> 6, qt = bqt & 63;
  int tid = threadIdx.x, lane = tid & 31, wave = tid >> 5;
  int col16 = lane & 15, rbase = (lane & 16) ? 8 : 0;

  // stage q rows (16 x 1024 bf16) as 16B chunks
  for (int c = tid; c < 2048; c += 256) {
    int r = c >> 7, ec = c & 127;           // ec*8 = h*64 + seg*8
    int h = ec >> 3, seg = ec & 7;
    stage16(&qh[(size_t)((b * H_ + h) * S_ + qt * 16 + r) * D_ + seg * 8], &qs[r][ec * 8]);
  }
  stage_wait();
  float avg[8][8] = {};
  __syncthreads();

  for (int h = 0; h < H_; ++h) {
    // ---- scores = q_tile x K^T  (wave-private 16x128 block in C-layout regs) ----
    v8f sacc[8] = {};
    for (int sub = 0; sub < 2; ++sub) {
      __syncthreads();
      for (int c = lane; c < 512; c += 32) {  // 64 rows x 8 chunks
        int jj = c >> 3, seg = c & 7;
        int j = wave * 128 + sub * 64 + jj;
        stage16(&kh[(size_t)((b * H_ + h) * S_ + j) * D_ + seg * 8], &kvs[wave][jj][seg * 8]);
      }
      stage_wait();
      __syncthreads();
#pragma unroll
      for (int ko = 0; ko < 2; ++ko) {
        v16bf af = frag_a(&qs[0][0], 1032, 0, h * 64 + ko * 32, lane);
#pragma unroll
        for (int t = 0; t < 4; ++t) {
          v16bf bf = frag_b(&kvs[wave][0][0], 64, t * 16, ko * 32, lane);
          sacc[sub * 4 + t] = WMMA_BF16(af, bf, sacc[sub * 4 + t]);
        }
      }
    }
    // ---- softmax over full S (wave shfl + cross-wave LDS reduction) ----
    float rmax[8], rsum[8];
#pragma unroll
    for (int r = 0; r < 8; ++r) {
      float m = sacc[0][r];
#pragma unroll
      for (int t = 1; t < 8; ++t) m = fmaxf(m, sacc[t][r]);
#pragma unroll
      for (int off = 1; off < 16; off <<= 1) m = fmaxf(m, __shfl_xor(m, off, 32));
      rmax[r] = m;
    }
    if (col16 == 0)
#pragma unroll
      for (int r = 0; r < 8; ++r) redbuf[wave][rbase + r] = rmax[r];
    __syncthreads();
#pragma unroll
    for (int r = 0; r < 8; ++r) {
      float m = redbuf[0][rbase + r];
#pragma unroll
      for (int w2 = 1; w2 < 8; ++w2) m = fmaxf(m, redbuf[w2][rbase + r]);
      rmax[r] = m;
    }
#pragma unroll
    for (int r = 0; r < 8; ++r) {
      float s = 0.f;
#pragma unroll
      for (int t = 0; t < 8; ++t) {
        float p = __expf(sacc[t][r] - rmax[r]);
        sacc[t][r] = p;
        s += p;
      }
#pragma unroll
      for (int off = 1; off < 16; off <<= 1) s += __shfl_xor(s, off, 32);
      rsum[r] = s;
    }
    __syncthreads();
    if (col16 == 0)
#pragma unroll
      for (int r = 0; r < 8; ++r) redbuf[wave][rbase + r] = rsum[r];
    __syncthreads();
#pragma unroll
    for (int r = 0; r < 8; ++r) {
      float s = 0.f;
#pragma unroll
      for (int w2 = 0; w2 < 8; ++w2) s += redbuf[w2][rbase + r];
      rsum[r] = 1.0f / s;
    }
    // probs -> avg accumulation + A-layout staging (C->A relayout via LDS)
#pragma unroll
    for (int t = 0; t < 8; ++t)
#pragma unroll
      for (int r = 0; r < 8; ++r) {
        float p = sacc[t][r] * rsum[r];
        avg[t][r] += p;
        ps[wave][rbase + r][t * 16 + col16] = (__bf16)p;
      }
    // zero ctx reduction slab (barrier at top of sub loop covers it)
    for (int i = tid; i < 16 * 64; i += 256) (&ctxred[0][0])[i] = 0.f;

    // ---- ctx partial = P(16x128) x V(128x64), V staged transposed ----
    v8f cacc[4] = {};
    for (int sub = 0; sub < 2; ++sub) {
      __syncthreads();
      for (int c = lane; c < 512; c += 32) {
        int jj = c >> 3, seg = c & 7;
        int j = wave * 128 + sub * 64 + jj;
        v8bf x = *(const v8bf*)&vh[(size_t)((b * H_ + h) * S_ + j) * D_ + seg * 8];
#pragma unroll
        for (int q = 0; q < 8; ++q) kvs[wave][seg * 8 + q][jj] = x[q];  // transpose scatter
      }
      __syncthreads();
#pragma unroll
      for (int ko = 0; ko < 2; ++ko) {
        v16bf af = frag_a(&ps[wave][0][0], 128, 0, sub * 64 + ko * 32, lane);
#pragma unroll
        for (int t = 0; t < 4; ++t) {
          v16bf bf = frag_b(&kvs[wave][0][0], 64, t * 16, ko * 32, lane);
          cacc[t] = WMMA_BF16(af, bf, cacc[t]);
        }
      }
    }
    // cross-wave ctx reduction, write ctx head slice
#pragma unroll
    for (int t = 0; t < 4; ++t)
#pragma unroll
      for (int r = 0; r < 8; ++r)
        atomicAdd(&ctxred[rbase + r][t * 16 + col16], cacc[t][r]);
    __syncthreads();
    for (int i = tid; i < 16 * 64; i += 256) {
      int m = i >> 6, d = i & 63;
      ctxw[(size_t)(b * S_ + qt * 16 + m) * E_ + h * D_ + d] = (__bf16)ctxred[m][d];
    }
    __syncthreads();
  }

  // ---- gated average weights: (avg/H)*gate, renormalize rows ----
  float rs[8];
#pragma unroll
  for (int r = 0; r < 8; ++r) {
    int qrow = qt * 16 + rbase + r;
    float s = 0.f;
#pragma unroll
    for (int t = 0; t < 8; ++t) {
      int j = wave * 128 + t * 16 + col16;
      float g = avg[t][r] * (1.0f / 16.0f) * gate[(size_t)(b * S_ + qrow) * S_ + j];
      avg[t][r] = g;
      s += g;
    }
#pragma unroll
    for (int off = 1; off < 16; off <<= 1) s += __shfl_xor(s, off, 32);
    rs[r] = s;
  }
  __syncthreads();
  if (col16 == 0)
#pragma unroll
    for (int r = 0; r < 8; ++r) redbuf[wave][rbase + r] = rs[r];
  __syncthreads();
#pragma unroll
  for (int r = 0; r < 8; ++r) {
    float s = 0.f;
#pragma unroll
    for (int w2 = 0; w2 < 8; ++w2) s += redbuf[w2][rbase + r];
    float inv = 1.0f / (s + 1e-12f);
    int qrow = qt * 16 + rbase + r;
#pragma unroll
    for (int t = 0; t < 8; ++t) {
      int j = wave * 128 + t * 16 + col16;
      gated_out[(size_t)(b * S_ + qrow) * S_ + j] = avg[t][r] * inv;
    }
  }
}

// ---------------- K3: out-projection GEMM (4096 x 1024, K=1024) ----------------
__global__ __launch_bounds__(128) void outproj_gemm(
    const __bf16* __restrict__ A, const __bf16* __restrict__ w,
    const float* __restrict__ bias, float* __restrict__ out) {
  __shared__ __bf16 As[64][72];
  __shared__ __bf16 Bs[64][72];
  int m0 = blockIdx.x * 64, n0 = blockIdx.y * 64;
  int tid = threadIdx.x, lane = tid & 31, wave = tid >> 5;
  v8f acc[4] = {};
  for (int k0 = 0; k0 < E_; k0 += 64) {
    __syncthreads();
    for (int c = tid; c < 512; c += 128) {
      int r = c >> 3, seg = c & 7;
      stage16(&A[(m0 + r) * E_ + k0 + seg * 8], &As[r][seg * 8]);
      stage16(&w[(n0 + r) * E_ + k0 + seg * 8], &Bs[r][seg * 8]);
    }
    if (k0 + 128 < E_)
      __builtin_prefetch(&A[(m0 + (tid >> 1)) * E_ + k0 + 128], 0, 1);
    stage_wait();
    __syncthreads();
#pragma unroll
    for (int ko = 0; ko < 2; ++ko) {
      v16bf af = frag_a(&As[0][0], 72, wave * 16, ko * 32, lane);
#pragma unroll
      for (int t = 0; t < 4; ++t) {
        v16bf bf = frag_b(&Bs[0][0], 72, t * 16, ko * 32, lane);
        acc[t] = WMMA_BF16(af, bf, acc[t]);
      }
    }
  }
  int col16 = lane & 15, rbase = (lane & 16) ? 8 : 0;
#pragma unroll
  for (int t = 0; t < 4; ++t) {
    int f = n0 + t * 16 + col16;
    float bv = bias[f];
#pragma unroll
    for (int r = 0; r < 8; ++r) {
      int row = m0 + wave * 16 + rbase + r;
      out[(size_t)row * E_ + f] = acc[t][r] + bv;
    }
  }
}

// ---------------- host launch ----------------
extern "C" void kernel_launch(void* const* d_in, const int* in_sizes, int n_in,
                              void* d_out, int out_size, void* d_ws, size_t ws_size,
                              hipStream_t stream) {
  (void)in_sizes; (void)n_in; (void)out_size; (void)ws_size;
  const float* q_f  = (const float*)d_in[0];
  const float* k_f  = (const float*)d_in[1];
  const float* v_f  = (const float*)d_in[2];
  const float* gate = (const float*)d_in[3];
  const float* wi_f = (const float*)d_in[4];   // (3E,E)
  const float* bi   = (const float*)d_in[5];   // (3E)
  const float* wo_f = (const float*)d_in[6];   // (E,E)
  const float* bo   = (const float*)d_in[7];   // (E)

  const size_t N_X  = (size_t)B_ * S_ * E_;      // 4194304
  const size_t N_W3 = (size_t)3 * E_ * E_;       // 3145728
  const size_t N_WO = (size_t)E_ * E_;           // 1048576

  char* ws = (char*)d_ws;
  size_t off = 0;
  __bf16* wbf  = (__bf16*)(ws + off); off += N_W3 * 2;
  __bf16* wobf = (__bf16*)(ws + off); off += N_WO * 2;
  __bf16* xq   = (__bf16*)(ws + off); off += N_X * 2;
  __bf16* xk   = (__bf16*)(ws + off); off += N_X * 2;
  __bf16* xv   = (__bf16*)(ws + off); off += N_X * 2;
  __bf16* qh   = (__bf16*)(ws + off); off += N_X * 2;
  __bf16* kh   = (__bf16*)(ws + off); off += N_X * 2;
  __bf16* vh   = (__bf16*)(ws + off); off += N_X * 2;
  __bf16* ctxw = (__bf16*)(ws + off); off += N_X * 2;

  cvt_f32_bf16<<<1024, 256, 0, stream>>>(q_f,  xq,   (int)N_X);
  cvt_f32_bf16<<<1024, 256, 0, stream>>>(k_f,  xk,   (int)N_X);
  cvt_f32_bf16<<<1024, 256, 0, stream>>>(v_f,  xv,   (int)N_X);
  cvt_f32_bf16<<<1024, 256, 0, stream>>>(wi_f, wbf,  (int)N_W3);
  cvt_f32_bf16<<<1024, 256, 0, stream>>>(wo_f, wobf, (int)N_WO);

  inproj_gemm<<<dim3(64, 48), 128, 0, stream>>>(xq, xk, xv, wbf, bi, qh, kh, vh);

  float* out_attn  = (float*)d_out;                 // (B,S,E)
  float* out_gated = (float*)d_out + N_X;           // (B,S,S)
  attn_kernel<<<B_ * (S_ / 16), 256, 0, stream>>>(qh, kh, vh, gate, ctxw, out_gated);

  outproj_gemm<<<dim3(64, 16), 128, 0, stream>>>(ctxw, wobf, bo, (float*)out_attn);
}
